// GCN_2_35184372089477
// MI455X (gfx1250) — compile-verified, matrix operand order checked by reading, not run on hardware
//
#include <hip/hip_runtime.h>
#include <hip/hip_bf16.h>

// CGCNN conv layer for MI455X (gfx1250, wave32, WMMA).
// k0 pack W(bf16, fragment-major)+zero stats -> k1 edge GEMM + BN stats ->
// k2 finalize scale/shift -> k3 recompute GEMM, apply BN, gate, bond_out,
// per-node 12-edge sum + node BN stats -> k4 node BN + softplus output.
// edge_index treated as int32 (JAX default demotes int64 without x64 mode).

#define NNODE 100000
#define NNBR  12
#define NEDGE (NNODE * NNBR)
#define DIM   192
#define ATOMF 64
#define EPSB  1e-5f

typedef __attribute__((ext_vector_type(16))) __bf16 v16bf;
typedef __attribute__((ext_vector_type(8)))  float  v8f;

// LDS carve (bytes)
#define WS_OFF   0        // __bf16 Ws[12*6*32*16]  = 73728 (fragment-major, contiguous)
#define ZS_OFF   73728    // __bf16 Zs[96][200]     = 38400 (aliased by gbuf in k3)
#define ST_OFF   112128   // stats scratch
#define SMEM_SZ  113664
#define ZSTR 200
#define GSTR 68

// ---- optional CDNA5 async global->LDS path (guarded; falls back cleanly) ----
#if defined(__has_builtin)
#if __has_builtin(__builtin_amdgcn_global_load_async_to_lds_b128) && \
    __has_builtin(__builtin_amdgcn_s_wait_asynccnt)
#define USE_ASYNC_LDS 1
#endif
#endif
#ifndef USE_ASYNC_LDS
#define USE_ASYNC_LDS 0
#endif

#if USE_ASYNC_LDS
typedef int v4i __attribute__((vector_size(16)));
typedef __attribute__((address_space(1))) v4i* g_v4i_p;  // global (AS1)
typedef __attribute__((address_space(3))) v4i* l_v4i_p;  // LDS (AS3)

__device__ __forceinline__ g_v4i_p to_global(const void* p) {
  return (g_v4i_p)(unsigned long long)(uintptr_t)p;
}
__device__ __forceinline__ l_v4i_p to_lds(const void* p) {
  // generic LDS address: low 32 bits are the LDS byte offset (ISA 10.2 aperture)
  return (l_v4i_p)(unsigned)(uintptr_t)p;
}
#endif

__device__ __forceinline__ float softplusf(float v) {
  return fmaxf(v, 0.f) + log1pf(expf(-fabsf(v)));
}
__device__ __forceinline__ float sigm(float v) { return 1.f / (1.f + expf(-v)); }

// cvt 4 fp32 -> 4 bf16, store 8B to LDS
__device__ __forceinline__ void st4bf(__bf16* d, float4 v) {
  union { __bf16 h[4]; uint2 u; } r;
  r.h[0] = (__bf16)v.x; r.h[1] = (__bf16)v.y;
  r.h[2] = (__bf16)v.z; r.h[3] = (__bf16)v.w;
  *(uint2*)d = r.u;
}

// A-fragment (16-bit 16xK, ISA 7.12.2): elems 0..7 -> K=half*8+e,
// elems 8..15 -> K=half*8+16+(e-8)
__device__ __forceinline__ v16bf load_afrag(const __bf16* row, int half, int kbase) {
  v16bf v;
  const int k0 = kbase + half * 8;
#pragma unroll
  for (int e = 0; e < 8; ++e) v[e] = row[k0 + e];
#pragma unroll
  for (int e = 0; e < 8; ++e) v[8 + e] = row[k0 + 16 + e];
  return v;
}

// B-fragment: fragment-major Ws, 32 contiguous bytes per (nt, s, lane)
__device__ __forceinline__ v16bf load_bfrag(const __bf16* Ws, int idx) {
  union { v16bf v; uint4 q[2]; } f;
  const uint4* p = (const uint4*)(Ws + idx * 16);
  f.q[0] = p[0];
  f.q[1] = p[1];
  return f.v;
}

// Stage W + gathered Z into LDS, then 16x192 x 192x192 GEMM per wave.
__device__ __forceinline__ void stage_and_gemm(
    const float* __restrict__ x, const int* __restrict__ dst,
    const float* __restrict__ eattr, const __bf16* __restrict__ Wb,
    __bf16* Ws, __bf16* Zs, int blk, int tid, v8f acc[12]) {
  // W: contiguous 73728B global -> LDS
#if USE_ASYNC_LDS
  for (int i = tid; i < 4608; i += 192)
    __builtin_amdgcn_global_load_async_to_lds_b128(
        to_global((const char*)Wb + i * 16), to_lds((char*)Ws + i * 16), 0, 0);
#else
  for (int i = tid; i < 4608; i += 192)
    ((uint4*)Ws)[i] = ((const uint4*)Wb)[i];
#endif

  // Z: 96 edges x 192 cols = [x[dst] | x[e/12] | edge_attr], fp32 -> bf16
  {
    const int row = tid >> 1;
    const int e   = blk * 96 + row;
    __bf16* zr = Zs + row * ZSTR;
    if ((tid & 1) == 0) {
      const float4* a = (const float4*)(x + (size_t)dst[e] * ATOMF);
      const float4* b = (const float4*)(x + (size_t)(e / NNBR) * ATOMF);
#pragma unroll
      for (int i = 0; i < 16; ++i) st4bf(zr + i * 4, a[i]);
#pragma unroll
      for (int i = 0; i < 8; ++i) st4bf(zr + 64 + i * 4, b[i]);
    } else {
      const float4* b = (const float4*)(x + (size_t)(e / NNBR) * ATOMF) + 8;
      const float4* c = (const float4*)(eattr + (size_t)e * ATOMF);
#pragma unroll
      for (int i = 0; i < 8; ++i) st4bf(zr + 96 + i * 4, b[i]);
#pragma unroll
      for (int i = 0; i < 16; ++i) st4bf(zr + 128 + i * 4, c[i]);
    }
  }
#if USE_ASYNC_LDS
  __builtin_amdgcn_s_wait_asynccnt(0);
#endif
  __syncthreads();

  const int lane  = tid & 31;
  const int wave  = tid >> 5;
  const int nlane = lane & 15;
  const int half  = lane >> 4;
  const __bf16* Arow = Zs + (wave * 16 + nlane) * ZSTR;

#pragma unroll
  for (int nt = 0; nt < 12; ++nt)
#pragma unroll
    for (int r = 0; r < 8; ++r) acc[nt][r] = 0.f;

#pragma unroll
  for (int s = 0; s < 6; ++s) {
    v16bf a = load_afrag(Arow, half, 32 * s);
#pragma unroll
    for (int nt = 0; nt < 12; ++nt) {
      v16bf b = load_bfrag(Ws, (nt * 6 + s) * 32 + lane);
      acc[nt] = __builtin_amdgcn_wmma_f32_16x16x32_bf16(
          false, a, false, b, (short)0, acc[nt], false, false);
    }
  }
}

// Pack W_core/W_filter/W_bond into fragment-major bf16:
// dest index i = (((nt*6)+s)*32 + lane)*16 + e  ->  (n, k) source element.
__global__ void k0_prep(const float* __restrict__ Wc, const float* __restrict__ Wf,
                        const float* __restrict__ Wbn, __bf16* __restrict__ Wb,
                        float* colS, float* colQ, float* nS, float* nQ) {
  int i = blockIdx.x * blockDim.x + threadIdx.x;
  if (i < DIM * DIM) {
    int e    = i & 15;
    int lane = (i >> 4) & 31;
    int sk   = (i >> 9) % 6;
    int nt   = i / 3072;
    int n    = nt * 16 + (lane & 15);
    int half = lane >> 4;
    int kk   = half * 8 + (e & 7) + ((e >> 3) << 4);
    int k    = sk * 32 + kk;
    const float* src = (n < 64) ? (Wc + n * DIM)
                     : (n < 128) ? (Wf + (n - 64) * DIM)
                                 : (Wbn + (n - 128) * DIM);
    Wb[i] = (__bf16)src[k];
  }
  if (i < DIM) { colS[i] = 0.f; colQ[i] = 0.f; }
  if (i < 64)  { nS[i] = 0.f;  nQ[i] = 0.f; }
}

__global__ __launch_bounds__(192) void k1_gemm_stats(
    const float* __restrict__ x, const int* __restrict__ dst,
    const float* __restrict__ eattr, const __bf16* __restrict__ Wb,
    float* colS, float* colQ) {
  extern __shared__ char smem[];
  __bf16* Ws = (__bf16*)(smem + WS_OFF);
  __bf16* Zs = (__bf16*)(smem + ZS_OFF);
  float*  lS = (float*)(smem + ST_OFF);        // [192]
  float*  lQ = (float*)(smem + ST_OFF + 768);  // [192]
  const int tid = threadIdx.x, blk = blockIdx.x;
  for (int i = tid; i < DIM; i += 192) { lS[i] = 0.f; lQ[i] = 0.f; }

  v8f acc[12];
  stage_and_gemm(x, dst, eattr, Wb, Ws, Zs, blk, tid, acc);

  const int lane = tid & 31, nlane = lane & 15;
#pragma unroll
  for (int nt = 0; nt < 12; ++nt) {
    float s = 0.f, q = 0.f;
#pragma unroll
    for (int r = 0; r < 8; ++r) { float v = acc[nt][r]; s += v; q += v * v; }
    atomicAdd(&lS[nt * 16 + nlane], s);
    atomicAdd(&lQ[nt * 16 + nlane], q);
  }
  __syncthreads();
  for (int i = tid; i < DIM; i += 192) {
    atomicAdd(&colS[i], lS[i]);
    atomicAdd(&colQ[i], lQ[i]);
  }
}

__global__ void k2_finalize(const float* colS, const float* colQ,
                            const float* g1, const float* be1,
                            const float* g2, const float* be2,
                            const float* g3, const float* be3,
                            float* scl, float* shf) {
  int n = blockIdx.x * blockDim.x + threadIdx.x;
  if (n >= DIM) return;
  // biases cancel in train-mode BN (shift both y and batch mean)
  float inv = 1.f / (float)NEDGE;
  float m   = colS[n] * inv;
  float var = colQ[n] * inv - m * m;
  float g, b;
  if (n < 64)       { g = g1[n];       b = be1[n]; }
  else if (n < 128) { g = g2[n - 64];  b = be2[n - 64]; }
  else              { g = g3[n - 128]; b = be3[n - 128]; }
  float sc = g * rsqrtf(var + EPSB);
  scl[n] = sc;
  shf[n] = b - m * sc;
}

__global__ __launch_bounds__(192) void k3_apply(
    const float* __restrict__ x, const int* __restrict__ dst,
    const float* __restrict__ eattr, const __bf16* __restrict__ Wb,
    const float* __restrict__ scl, const float* __restrict__ shf,
    float* __restrict__ bondOut, float* __restrict__ nodeNbr,
    float* nS, float* nQ) {
  extern __shared__ char smem[];
  __bf16* Ws   = (__bf16*)(smem + WS_OFF);
  __bf16* Zs   = (__bf16*)(smem + ZS_OFF);
  float*  gbuf = (float*)(smem + ZS_OFF);       // aliases Zs after GEMM
  float*  cS   = (float*)(smem + ST_OFF);       // [64]
  float*  cQ   = (float*)(smem + ST_OFF + 256); // [64]
  const int tid = threadIdx.x, blk = blockIdx.x;
  for (int i = tid; i < 64; i += 192) { cS[i] = 0.f; cQ[i] = 0.f; }

  v8f acc[12];
  stage_and_gemm(x, dst, eattr, Wb, Ws, Zs, blk, tid, acc);

  const int lane = tid & 31, wave = tid >> 5;
  const int nlane = lane & 15, half = lane >> 4;
  const int eBase = blk * 96 + wave * 16;

  // bond_out = softplus(edge_attr + bn3(bond)), cols 128..191
#pragma unroll
  for (int nt = 8; nt < 12; ++nt) {
    int n = nt * 16 + nlane;
    float sc = scl[n], sh = shf[n];
    int cb = n - 128;
#pragma unroll
    for (int r = 0; r < 8; ++r) {
      int e = eBase + r + 8 * half;
      float bnv = acc[nt][r] * sc + sh;
      bondOut[(size_t)e * ATOMF + cb] =
          softplusf(eattr[(size_t)e * ATOMF + cb] + bnv);
    }
  }
  __syncthreads();  // all waves done reading Zs; reuse as gbuf

  // gated = sigmoid(bn1(core)) * softplus(bn2(filter))   (reference quirk kept)
#pragma unroll
  for (int nt = 0; nt < 4; ++nt) {
    int c = nt * 16 + nlane;
    float scF = scl[c],      shF = shf[c];
    float scC = scl[c + 64], shC = shf[c + 64];
#pragma unroll
    for (int r = 0; r < 8; ++r) {
      int m = r + 8 * half;
      float af = acc[nt][r]     * scF + shF;
      float ac = acc[nt + 4][r] * scC + shC;
      gbuf[(wave * 16 + m) * GSTR + c] = sigm(af) * softplusf(ac);
    }
  }
  __syncthreads();

  // block holds exactly 8 nodes x 12 edges -> segment sum without atomics
  for (int idx = tid; idx < 8 * 64; idx += 192) {
    int ln = idx >> 6, c = idx & 63;
    float s = 0.f;
#pragma unroll
    for (int j = 0; j < NNBR; ++j) s += gbuf[(ln * NNBR + j) * GSTR + c];
    nodeNbr[(size_t)(blk * 8 + ln) * ATOMF + c] = s;
    atomicAdd(&cS[c], s);
    atomicAdd(&cQ[c], s * s);
  }
  __syncthreads();
  for (int i = tid; i < 64; i += 192) {
    atomicAdd(&nS[i], cS[i]);
    atomicAdd(&nQ[i], cQ[i]);
  }
}

__global__ void k4_out(const float* __restrict__ x, const float* __restrict__ nodeNbr,
                       const float* nS, const float* nQ,
                       const float* g4, const float* be4,
                       float* __restrict__ outNode) {
  size_t i = (size_t)blockIdx.x * blockDim.x + threadIdx.x;
  if (i >= (size_t)NNODE * ATOMF) return;
  int c = (int)(i & 63);
  float inv = 1.f / (float)NNODE;
  float m   = nS[c] * inv;
  float var = nQ[c] * inv - m * m;
  float sc  = g4[c] * rsqrtf(var + EPSB);
  float sh  = be4[c] - m * sc;
  outNode[i] = softplusf(x[i] + nodeNbr[i] * sc + sh);
}

extern "C" void kernel_launch(void* const* d_in, const int* in_sizes, int n_in,
                              void* d_out, int out_size, void* d_ws, size_t ws_size,
                              hipStream_t stream) {
  (void)in_sizes; (void)n_in; (void)out_size; (void)ws_size;
  const float* x     = (const float*)d_in[0];
  const int*   ei    = (const int*)d_in[1];
  const int*   dst   = ei + NEDGE;           // edge_index[1]
  const float* eattr = (const float*)d_in[2];
  const float* Wc    = (const float*)d_in[3];
  const float* Wf    = (const float*)d_in[5];
  const float* Wbn   = (const float*)d_in[7];
  const float* g1  = (const float*)d_in[9];  const float* be1 = (const float*)d_in[10];
  const float* g2  = (const float*)d_in[11]; const float* be2 = (const float*)d_in[12];
  const float* g3  = (const float*)d_in[13]; const float* be3 = (const float*)d_in[14];
  const float* g4  = (const float*)d_in[15]; const float* be4 = (const float*)d_in[16];

  char* ws = (char*)d_ws;
  __bf16* Wb      = (__bf16*)(ws + 0);        // 192*192*2 = 73728 B (fragment-major)
  float*  colS    = (float*)(ws + 73728);     // [192]
  float*  colQ    = (float*)(ws + 74496);     // [192]
  float*  scl     = (float*)(ws + 75264);     // [192]
  float*  shf     = (float*)(ws + 76032);     // [192]
  float*  nS      = (float*)(ws + 76800);     // [64]
  float*  nQ      = (float*)(ws + 77056);     // [64]
  float*  nodeNbr = (float*)(ws + 77824);     // N*64 f32 = 25.6 MB

  float* outNode = (float*)d_out;
  float* outBond = outNode + (size_t)NNODE * ATOMF;

  k0_prep<<<(DIM * DIM + 255) / 256, 256, 0, stream>>>(Wc, Wf, Wbn, Wb, colS, colQ, nS, nQ);
  k1_gemm_stats<<<NEDGE / 96, 192, SMEM_SZ, stream>>>(x, dst, eattr, Wb, colS, colQ);
  k2_finalize<<<1, 192, 0, stream>>>(colS, colQ, g1, be1, g2, be2, g3, be3, scl, shf);
  k3_apply<<<NEDGE / 96, 192, SMEM_SZ, stream>>>(x, dst, eattr, Wb, scl, shf,
                                                 outBond, nodeNbr, nS, nQ);
  k4_out<<<((size_t)NNODE * ATOMF + 255) / 256, 256, 0, stream>>>(x, nodeNbr, nS, nQ,
                                                                  g4, be4, outNode);
}
